// basicUNIT_54348516164066
// MI455X (gfx1250) — matrix operand order repaired and checked by reading, not compiled, not run on hardware
//
#include <hip/hip_runtime.h>

// ---------------------------------------------------------------------------
// Fused cosine-sim retrieval for MI455X (gfx1250, wave32):
//   1) prep_norm : normalize each row by 1/||row|| and store bf16
//                  (cosine sim == plain dot of normalized operands)
//   2) mask_rows : zero the bf16 memory rows at mask_idx (dot -> 0, exactly
//                  reproducing the reference's column masking)
//   3) topk_part : bf16 WMMA GEMM, 32 query rows/block (two A tiles share
//      each B fragment -> 1 b128 load per WMMA); cross-tile double-buffered
//      B staging keeps a full tile of loads in flight behind the WMMAs;
//      branchless register-cached-threshold fast path (no LDS, one ballot);
//      intra-block merge -> per-(row,half) top-32 candidates in global ws.
//   4) finalize  : merge 2x32 -> top-32, softmax(relu), fp32 gather-combine.
// ---------------------------------------------------------------------------

typedef __attribute__((ext_vector_type(16))) __bf16 v16bf;
typedef __attribute__((ext_vector_type(8)))  __bf16 v8bf;
typedef __attribute__((ext_vector_type(8)))  float  v8f;

#define MEMSZ  50000
#define DIM    256
#define BATCH  4096
#define NMASK  5000
#define KSEL   32
#define NTILES (MEMSZ / 16)   // 3125 column tiles of 16 memory rows
#define NWAVES 8
#define ROWS   32             // query rows per block (two 16-row A tiles)

__device__ __forceinline__ unsigned short f2bf(float f) {
    unsigned int u = __float_as_uint(f);
    unsigned int r = u + 0x7FFFu + ((u >> 16) & 1u);  // RNE
    return (unsigned short)(r >> 16);
}

__device__ __forceinline__ unsigned wave_ballot(bool p) {
#if __has_builtin(__builtin_amdgcn_ballot_w32)
    return __builtin_amdgcn_ballot_w32(p);
#else
    return (unsigned)__ballot(p);
#endif
}

// ---- prep: normalize row to unit length, store bf16 -----------------------
__global__ void prep_norm(const float* __restrict__ src,
                          unsigned short* __restrict__ dstb, int nrows) {
    int wv = threadIdx.x >> 5;
    int ln = threadIdx.x & 31;
    int r  = blockIdx.x * NWAVES + wv;
    if (r >= nrows) return;
    const float* row = src + (size_t)r * DIM;
    float v[8];
    float ss = 0.f;
#pragma unroll
    for (int c = 0; c < 8; ++c) { v[c] = row[ln + 32 * c]; ss += v[c] * v[c]; }
#pragma unroll
    for (int off = 16; off; off >>= 1) ss += __shfl_xor(ss, off, 32);
    const float rinv = rsqrtf(ss);
    unsigned short* drow = dstb + (size_t)r * DIM;
#pragma unroll
    for (int c = 0; c < 8; ++c) drow[ln + 32 * c] = f2bf(v[c] * rinv);
}

// ---- zero the bf16 rows of masked memory slots ----------------------------
__global__ void mask_rows(const int* __restrict__ mask_idx,
                          unsigned short* __restrict__ memb, int n) {
    int wv = threadIdx.x >> 5;
    int ln = threadIdx.x & 31;
    int i  = blockIdx.x * NWAVES + wv;
    if (i >= n) return;
    uint4 z = {0u, 0u, 0u, 0u};
    unsigned short* row = memb + (size_t)mask_idx[i] * DIM;
    *(uint4*)(row + ln * 8) = z;           // 32 lanes x 16 B = 512 B row
}

// load A fragment: slots 0-7 <- p[0..7], slots 8-15 <- p[8..15]
__device__ __forceinline__ v16bf ld_a_frag(const unsigned short* p) {
    v8bf lo = *(const v8bf*)p;
    v8bf hi = *(const v8bf*)(p + 8);
    v16bf r;
#pragma unroll
    for (int i = 0; i < 8; ++i) { r[i] = lo[i]; r[i + 8] = hi[i]; }
    return r;
}

// threshold-guarded cooperative replace-min insertion into this wave's lists
__device__ __forceinline__ void topk_insert(float* sval, unsigned short* sidx,
                                            float* sthr, int ln,
                                            float myval, int myrow, int mycol) {
    bool pend = true;
    while (true) {
        float    thr = sthr[myrow];
        unsigned msk = wave_ballot(pend && (myval > thr));
        if (msk == 0) break;
        int leader = __ffs(msk) - 1;
        int   lrow = __shfl(myrow, leader, 32);
        float lval = __shfl(myval, leader, 32);
        int   lcol = __shfl(mycol, leader, 32);
        if (ln == leader) pend = false;
        float* vrow = sval + lrow * KSEL;
        float ev = vrow[ln];
        float mv = ev; int mlane = ln;
#pragma unroll
        for (int off = 16; off; off >>= 1) {
            float ov = __shfl_xor(mv, off, 32);
            int   ol = __shfl_xor(mlane, off, 32);
            if (ov < mv || (ov == mv && ol < mlane)) { mv = ov; mlane = ol; }
        }
        if (lval > mv) {
            if (ln == mlane) {
                vrow[ln] = lval;
                sidx[lrow * KSEL + ln] = (unsigned short)lcol;
            }
            float ev2 = (ln == mlane) ? lval : ev;
#pragma unroll
            for (int off = 16; off; off >>= 1)
                ev2 = fminf(ev2, __shfl_xor(ev2, off, 32));
            if (ln == 0) sthr[lrow] = ev2;
        }
    }
}

// issue all 16 b128 loads (8 fragments) for one column tile
__device__ __forceinline__ void tile_load(v16bf (&b)[8],
                                          const unsigned short* __restrict__ memb,
                                          int n, int hioff) {
    const unsigned short* bp = memb + (size_t)n * DIM + hioff;
#pragma unroll
    for (int c = 0; c < 8; ++c) b[c] = *(const v16bf*)(bp + c * 32);
}

// ---- main GEMM + partial top-k kernel -------------------------------------
__launch_bounds__(256, 1)
__global__ void topk_part(const unsigned short* __restrict__ memb,
                          const unsigned short* __restrict__ encb,
                          float* __restrict__ pval,
                          int* __restrict__ pidx) {
    __shared__ float          s_val[NWAVES][ROWS][KSEL];  // 32 KB
    __shared__ unsigned short s_idx[NWAVES][ROWS][KSEL];  // 16 KB
    __shared__ float          s_thr[NWAVES][ROWS];        //  1 KB

    const int tid    = threadIdx.x;
    const int wv     = tid >> 5;
    const int ln     = tid & 31;
    const int rowblk = blockIdx.x >> 1;
    const int half   = blockIdx.x & 1;
    const int m0     = rowblk * ROWS;
    const int tbeg   = (half * NTILES) / 2;
    const int tend   = ((half + 1) * NTILES) / 2;

    for (int i = tid; i < NWAVES * ROWS * KSEL; i += 256) {
        ((float*)s_val)[i]          = -INFINITY;
        ((unsigned short*)s_idx)[i] = 0;
    }
    for (int i = tid; i < NWAVES * ROWS; i += 256) ((float*)s_thr)[i] = -INFINITY;
    __syncthreads();

    const int lrow16 = ln & 15;
    const int hi     = ln >> 4;   // 0: lanes 0-15, 1: lanes 16-31
    const int hioff  = hi * 16;

    // preload TWO A tiles (rows m0..+15, m0+16..+31) x K=256 into registers
    v16bf a0[8], a1[8];
    {
        const unsigned short* ap0 = encb + (size_t)(m0 + lrow16) * DIM + hi * 16;
        const unsigned short* ap1 = encb + (size_t)(m0 + 16 + lrow16) * DIM + hi * 16;
#pragma unroll
        for (int c = 0; c < 8; ++c) {
            a0[c] = ld_a_frag(ap0 + c * 32);
            a1[c] = ld_a_frag(ap1 + c * 32);
        }
    }

    float* my_val = &s_val[wv][0][0];
    unsigned short* my_idx = &s_idx[wv][0][0];
    float* my_thr = &s_thr[wv][0];

    // 8 conservative cached thresholds: min over the two rows sharing slot v.
    // s_thr only increases, so a stale/low cache is correct (extra slow-path
    // entries at worst).
    float thrC[8];
#pragma unroll
    for (int i = 0; i < 8; ++i) thrC[i] = -INFINITY;

    // WMMA + top-k for one tile whose fragments are already staged in b[]
    auto tile_process = [&](const v16bf (&b)[8], int n) {
        v8f acc0 = {};
        v8f acc1 = {};
#pragma unroll
        for (int c = 0; c < 8; ++c) {
            acc0 = __builtin_amdgcn_wmma_f32_16x16x32_bf16(
                false, a0[c], false, b[c], (short)0, acc0, false, false);
            acc1 = __builtin_amdgcn_wmma_f32_16x16x32_bf16(
                false, a1[c], false, b[c], (short)0, acc1, false, false);
        }
        // branchless fast path: bitwise OR accumulation, one ballot,
        // no LDS in the common (no-insertion) case
        int any = 0;
#pragma unroll
        for (int v = 0; v < 8; ++v) {
            float m = fmaxf(acc0[v], acc1[v]);
            any |= (m > thrC[v]) ? 1 : 0;
        }
        if (wave_ballot(any != 0) != 0) {     // uniform slow path (rare)
#pragma unroll 1
            for (int v = 0; v < 8; ++v)
                topk_insert(my_val, my_idx, my_thr, ln,
                            acc0[v], v + hi * 8, n);
#pragma unroll 1
            for (int v = 0; v < 8; ++v)
                topk_insert(my_val, my_idx, my_thr, ln,
                            acc1[v], 16 + v + hi * 8, n);
#pragma unroll 1
            for (int v = 0; v < 8; ++v)
                thrC[v] = fminf(my_thr[v + hi * 8], my_thr[16 + v + hi * 8]);
        }
    };

    // -------- software-pipelined tile loop (double-buffered B) -------------
    v16bf bA[8], bB[8];
    int t = tbeg + wv;
    if (t < tend) tile_load(bA, memb, t * 16 + lrow16, hioff);
    while (t < tend) {
        const int t1 = t + NWAVES;
        if (t1 < tend) tile_load(bB, memb, t1 * 16 + lrow16, hioff);
        tile_process(bA, t * 16 + lrow16);
        if (t1 >= tend) break;
        const int t2 = t1 + NWAVES;
        if (t2 < tend) tile_load(bA, memb, t2 * 16 + lrow16, hioff);
        tile_process(bB, t1 * 16 + lrow16);
        t = t2;
    }
    __syncthreads();

    // -------- intra-block merge: 8x32 -> top-32 per row -> global ----------
    for (int rr = 0; rr < 4; ++rr) {
        const int r = wv + rr * 8;
        float cv[NWAVES];
        int   ci[NWAVES];
#pragma unroll
        for (int j = 0; j < NWAVES; ++j) {
            cv[j] = s_val[j][r][ln];
            ci[j] = (int)s_idx[j][r][ln];
        }
        const size_t obase = ((size_t)(m0 + r) * 2 + half) * KSEL;
        for (int t2 = 0; t2 < KSEL; ++t2) {
            float bv = cv[0]; int bj = 0;
#pragma unroll
            for (int j = 1; j < NWAVES; ++j)
                if (cv[j] > bv) { bv = cv[j]; bj = j; }
            float rv = bv; int rl = ln;
#pragma unroll
            for (int off = 16; off; off >>= 1) {
                float ov = __shfl_xor(rv, off, 32);
                int   ol = __shfl_xor(rl, off, 32);
                if (ov > rv || (ov == rv && ol < rl)) { rv = ov; rl = ol; }
            }
            if (ln == rl) {
                pval[obase + t2] = bv;
                pidx[obase + t2] = ci[bj];
                cv[bj] = -INFINITY;
            }
        }
    }
}

// ---- finalize: merge halves, softmax(relu), fp32 gather-combine -----------
__launch_bounds__(256, 2)
__global__ void finalize(const float* __restrict__ pval,
                         const int* __restrict__ pidx,
                         const float* __restrict__ memf,
                         float* __restrict__ out) {
    __shared__ float s_w[NWAVES][KSEL];
    __shared__ int   s_i[NWAVES][KSEL];

    const int wv = threadIdx.x >> 5;
    const int ln = threadIdx.x & 31;
    const int r  = blockIdx.x * NWAVES + wv;   // one wave per query row

    float cv[2];
    int   ci[2];
    cv[0] = pval[(size_t)r * 64 + ln];
    cv[1] = pval[(size_t)r * 64 + 32 + ln];
    ci[0] = pidx[(size_t)r * 64 + ln];
    ci[1] = pidx[(size_t)r * 64 + 32 + ln];

    for (int t = 0; t < KSEL; ++t) {
        float bv = cv[0]; int bj = 0;
        if (cv[1] > bv) { bv = cv[1]; bj = 1; }
        float rv = bv; int rl = ln;
#pragma unroll
        for (int off = 16; off; off >>= 1) {
            float ov = __shfl_xor(rv, off, 32);
            int   ol = __shfl_xor(rl, off, 32);
            if (ov > rv || (ov == rv && ol < rl)) { rv = ov; rl = ol; }
        }
        if (ln == rl) {
            s_w[wv][t] = bv;
            s_i[wv][t] = ci[bj];
            cv[bj] = -INFINITY;
        }
    }

    // softmax(relu(top-k)) over the 32 selected entries
    float sv  = s_w[wv][ln];
    float e   = __expf(fmaxf(sv, 0.0f));
    float sum = e;
#pragma unroll
    for (int off = 16; off; off >>= 1) sum += __shfl_xor(sum, off, 32);
    s_w[wv][ln] = e / sum;

    // fp32 weighted gather-combine
    float accd[8];
#pragma unroll
    for (int c = 0; c < 8; ++c) accd[c] = 0.0f;
    for (int j = 0; j < KSEL; ++j) {
        const float  w  = s_w[wv][j];
        const float* mr = memf + (size_t)s_i[wv][j] * DIM;
#pragma unroll
        for (int c = 0; c < 8; ++c) accd[c] += w * mr[ln + 32 * c];
    }
    float* orow = out + (size_t)r * DIM;
#pragma unroll
    for (int c = 0; c < 8; ++c) orow[ln + 32 * c] = accd[c];
}

// ---------------------------------------------------------------------------
#define OFF_ENCB  25600000u                  // after memb (50000*256*2)
#define OFF_PVAL  27697152u                  // after encb (4096*256*2)
#define OFF_PIDX  (OFF_PVAL + 1048576u)      // after pval (4096*64*4)

extern "C" void kernel_launch(void* const* d_in, const int* in_sizes, int n_in,
                              void* d_out, int out_size, void* d_ws, size_t ws_size,
                              hipStream_t stream) {
    const float* enc  = (const float*)d_in[0];   // [4096, 256] fp32
    const float* mem  = (const float*)d_in[1];   // [50000, 256] fp32
    const int*   mask = (const int*)d_in[2];     // [5000] int32
    // d_in[3] = k (compile-time constant 32)

    char* ws = (char*)d_ws;
    unsigned short* memb = (unsigned short*)ws;
    unsigned short* encb = (unsigned short*)(ws + OFF_ENCB);
    float*          pval = (float*)(ws + OFF_PVAL);
    int*            pidx = (int*)(ws + OFF_PIDX);
    float*          out  = (float*)d_out;

    prep_norm<<<MEMSZ / NWAVES, 256, 0, stream>>>(mem, memb, MEMSZ);
    prep_norm<<<BATCH / NWAVES, 256, 0, stream>>>(enc, encb, BATCH);
    mask_rows<<<NMASK / NWAVES, 256, 0, stream>>>(mask, memb, NMASK);
    topk_part<<<(BATCH / ROWS) * 2, 256, 0, stream>>>(memb, encb, pval, pidx);
    finalize<<<BATCH / NWAVES, 256, 0, stream>>>(pval, pidx, mem, out);
}
